// GNN_Autoencoder_circle_87909390615183
// MI455X (gfx1250) — matrix operand order repaired and checked by reading, not compiled, not run on hardware
//
#include <hip/hip_runtime.h>

// ---------------------------------------------------------------------------
// GNN autoencoder on 17-node circle graphs, CDNA5 (gfx1250, wave32, WMMA).
//
//  * degree==3 everywhere -> GCN norm is the constant 1/3; edge_index unused.
//  * fc + GCN1 weight fuse: W1eff[i] = fc_w[:,8i:8i+8] @ w1  (17 x [32x16]),
//    c1[i] = fc_b[8i:8i+8] @ w1, precomputed once into d_ws as f16 B-frags.
//  * all GEMMs run on v_wmma_f32_16x16x32_f16 (f16 in, f32 accumulate).
//  * one wave32 per 16 graphs; all aggregation loops are exec-uniform.
// ---------------------------------------------------------------------------

typedef __attribute__((ext_vector_type(16))) _Float16 v16h;
typedef __attribute__((ext_vector_type(8)))  _Float16 v8h;
typedef __attribute__((ext_vector_type(4)))  _Float16 v4h;
typedef __attribute__((ext_vector_type(8)))  float    v8f;

#define GNN_NN   17
#define TILE_G   16                  // graphs per wave
#define NROWS    (GNN_NN * TILE_G)   // 272 node-rows per wave
#define LDP      32                  // bufPre row stride (halves)
#define LDA      40                  // bufA row stride (halves); padded for banks

// d_ws layout (bytes); everything 32B aligned
#define WS_W1EFF 0        // 17 frags * 32 lanes * 16 halves = 17408 B
#define WS_W2F   17408    // 2 frags                          =  2048 B
#define WS_W3F   19456    // 2 frags                          =  2048 B
#define WS_C1    21504    // 17*16 floats                     =  1088 B

// ---------------------------------------------------------------------------
// Prep kernel: build f16 WMMA B-fragments for all weights.
// B-frag layout (16-bit B, 32x16, per ISA): lane l holds column n = l&15,
// elements e=0..7 are K = 8*(l>>4) + e, e=8..15 are K = 16 + 8*(l>>4) + (e-8).
// ---------------------------------------------------------------------------
__global__ __launch_bounds__(256) void gnn_prep_kernel(
    const float* __restrict__ fc_w,  // [32,136]
    const float* __restrict__ fc_b,  // [136]
    const float* __restrict__ w1,    // [8,16]
    const float* __restrict__ w2,    // [16,32]
    const float* __restrict__ w3,    // [32,32]
    _Float16* __restrict__ w1eff, _Float16* __restrict__ w2f,
    _Float16* __restrict__ w3f, float* __restrict__ c1)
{
  const int t = threadIdx.x;

  // W1eff[i] (32x16) = fc_w[:, 8i:8i+8] @ w1, stored as B-frags
  for (int idx = t; idx < 17 * 32 * 16; idx += 256) {
    const int e    = idx & 15;
    const int lane = (idx >> 4) & 31;
    const int i    = idx >> 9;
    const int n    = lane & 15;
    const int k    = ((lane >> 4) << 3) + (e & 7) + ((e >> 3) << 4);   // 0..31
    float acc = 0.f;
#pragma unroll
    for (int q = 0; q < 8; ++q)
      acc += fc_w[k * 136 + i * 8 + q] * w1[q * 16 + n];
    w1eff[idx] = (_Float16)acc;
  }

  // w2 (16x32) -> two B-frags, K zero-padded 16->32
  for (int idx = t; idx < 2 * 32 * 16; idx += 256) {
    const int e    = idx & 15;
    const int lane = (idx >> 4) & 31;
    const int c    = idx >> 9;                 // N-tile 0/1
    const int n    = c * 16 + (lane & 15);
    const int k    = ((lane >> 4) << 3) + (e & 7) + ((e >> 3) << 4);
    w2f[idx] = (_Float16)((k < 16) ? w2[k * 32 + n] : 0.f);
  }

  // w3 (32x32) -> two B-frags
  for (int idx = t; idx < 2 * 32 * 16; idx += 256) {
    const int e    = idx & 15;
    const int lane = (idx >> 4) & 31;
    const int c    = idx >> 9;
    const int n    = c * 16 + (lane & 15);
    const int k    = ((lane >> 4) << 3) + (e & 7) + ((e >> 3) << 4);
    w3f[idx] = (_Float16)w3[k * 32 + n];
  }

  // c1[i][n] = fc_b[8i:8i+8] @ w1  (pre-aggregation bias of fused layer 1)
  for (int idx = t; idx < 17 * 16; idx += 256) {
    const int n = idx & 15;
    const int i = idx >> 4;
    float acc = 0.f;
#pragma unroll
    for (int q = 0; q < 8; ++q) acc += fc_b[i * 8 + q] * w1[q * 16 + n];
    c1[idx] = acc;
  }
}

// ---------------------------------------------------------------------------
// Ring aggregation, layer-1 (F=16): relu((h[i-1]+h[i]+h[i+1])/3 + b1).
// Uniform counted loop; lane -> feature-group (lane&3)*4 of the 16 real
// features; the K-pad (cols 16..31) zero store is folded into each iteration.
// ---------------------------------------------------------------------------
__device__ __forceinline__ void aggregate16(const _Float16* pre, _Float16* dst,
                                            const float* __restrict__ bias,
                                            int lane)
{
  const int f  = (lane & 3) * 4;      // 0,4,8,12  (< 16 always)
  const int nb = lane >> 2;           // row sub-offset 0..7
  float bv[4];
#pragma unroll
  for (int j = 0; j < 4; ++j) bv[j] = bias[f + j];
  const v4h zpad = {};
#pragma unroll 2
  for (int k = 0; k < NROWS / 8; ++k) {            // 34 uniform iterations
    const int n  = nb + 8 * k;                     // node row (n = i*16+g)
    const int i  = n >> 4, g = n & 15;
    const int ip = (i == 0) ? (GNN_NN - 1) : i - 1;
    const int in = (i == GNN_NN - 1) ? 0 : i + 1;
    v4h a = *(const v4h*)(pre + n * LDP + f);
    v4h b = *(const v4h*)(pre + (ip * 16 + g) * LDP + f);
    v4h c = *(const v4h*)(pre + (in * 16 + g) * LDP + f);
    v4h o;
#pragma unroll
    for (int j = 0; j < 4; ++j) {
      float s = ((float)a[j] + (float)b[j] + (float)c[j]) * (1.0f / 3.0f) + bv[j];
      o[j] = (_Float16)fmaxf(s, 0.0f);
    }
    *(v4h*)(dst + n * LDA + f)      = o;
    *(v4h*)(dst + n * LDA + f + 16) = zpad;        // zero K-pad 16..31
  }
}

// Ring aggregation, F=32, relu; uniform counted loop.
__device__ __forceinline__ void aggregate32(const _Float16* pre, _Float16* dst,
                                            const float* __restrict__ bias,
                                            int lane)
{
  const int f  = (lane & 7) * 4;      // 0..28
  const int nb = lane >> 3;           // 0..3
  float bv[4];
#pragma unroll
  for (int j = 0; j < 4; ++j) bv[j] = bias[f + j];
#pragma unroll 2
  for (int k = 0; k < NROWS / 4; ++k) {            // 68 uniform iterations
    const int n  = nb + 4 * k;
    const int i  = n >> 4, g = n & 15;
    const int ip = (i == 0) ? (GNN_NN - 1) : i - 1;
    const int in = (i == GNN_NN - 1) ? 0 : i + 1;
    v4h a = *(const v4h*)(pre + n * LDP + f);
    v4h b = *(const v4h*)(pre + (ip * 16 + g) * LDP + f);
    v4h c = *(const v4h*)(pre + (in * 16 + g) * LDP + f);
    v4h o;
#pragma unroll
    for (int j = 0; j < 4; ++j) {
      float s = ((float)a[j] + (float)b[j] + (float)c[j]) * (1.0f / 3.0f) + bv[j];
      o[j] = (_Float16)fmaxf(s, 0.0f);
    }
    *(v4h*)(dst + n * LDA + f) = o;
  }
}

// Dense layer on 272 node-rows: 17 M-tiles x (two N=16 WMMAs), K=32.
__device__ __forceinline__ void dense_layer(const _Float16* srcA,
                                            _Float16* dstPre,
                                            v16h wlo, v16h whi, int lane)
{
  const int n16 = lane & 15;
  const int hi  = lane >> 4;
  for (int t = 0; t < GNN_NN; ++t) {
    // A-frag (16-bit A 16x32 ISA layout): elems 0..7 -> K=8hi.., 8..15 -> 16+8hi..
    const _Float16* ar = srcA + (size_t)(t * 16 + n16) * LDA;
    v8h p0 = *(const v8h*)(ar + hi * 8);
    v8h p1 = *(const v8h*)(ar + 16 + hi * 8);
    v16h a;
#pragma unroll
    for (int j = 0; j < 8; ++j) { a[j] = p0[j]; a[j + 8] = p1[j]; }
    v8f zc = {};
    v8f d0 = __builtin_amdgcn_wmma_f32_16x16x32_f16(false, a, false, wlo,
                                                    (short)0, zc, false, false);
    v8f d1 = __builtin_amdgcn_wmma_f32_16x16x32_f16(false, a, false, whi,
                                                    (short)0, zc, false, false);
    // D layout: lane elem j -> row M = j + 8*hi, col N = lane&15
    _Float16* dst = dstPre + (size_t)(t * 16 + hi * 8) * LDP + n16;
#pragma unroll
    for (int j = 0; j < 8; ++j) {
      dst[j * LDP]      = (_Float16)d0[j];
      dst[j * LDP + 16] = (_Float16)d1[j];
    }
  }
}

// ---------------------------------------------------------------------------
// Main kernel: one wave32 per 16 graphs.
// ---------------------------------------------------------------------------
__global__ __launch_bounds__(32) void gnn_main_kernel(
    const float* __restrict__ z,
    const float* __restrict__ b1, const float* __restrict__ b2,
    const float* __restrict__ b3,
    const _Float16* __restrict__ w1eff, const _Float16* __restrict__ w2f,
    const _Float16* __restrict__ w3f, const float* __restrict__ c1,
    float* __restrict__ out)
{
  __shared__ _Float16 bufPre[NROWS * LDP];   // pre-aggregation features (f16)
  __shared__ _Float16 bufA[NROWS * LDA];     // A-matrix source rows (f16, K=32)

  const int lane = threadIdx.x & 31;
  const int n16  = lane & 15;
  const int hi   = lane >> 4;
  const int g0   = blockIdx.x * TILE_G;

  // z-tile A-frag (16 graphs x 32 feats), f32 -> f16, held in regs all of L1
  v16h za;
  {
    const float* zr = z + (size_t)(g0 + n16) * 32;
#pragma unroll
    for (int j = 0; j < 8; ++j) {
      za[j]     = (_Float16)zr[hi * 8 + j];        // K = 8hi..8hi+7
      za[j + 8] = (_Float16)zr[16 + hi * 8 + j];   // K = 16+8hi..
    }
  }

  // weight B-frags for layers 2/3 pinned in VGPRs
  v16h w2lo = *(const v16h*)(w2f + (size_t)(0 * 32 + lane) * 16);
  v16h w2hi = *(const v16h*)(w2f + (size_t)(1 * 32 + lane) * 16);
  v16h w3lo = *(const v16h*)(w3f + (size_t)(0 * 32 + lane) * 16);
  v16h w3hi = *(const v16h*)(w3f + (size_t)(1 * 32 + lane) * 16);

  // ---- fused fc + GCN1 pre-aggregation: per position i, H_i = Z @ W1eff[i] + c1[i]
#pragma unroll 1
  for (int i = 0; i < GNN_NN; ++i) {
    v16h bw = *(const v16h*)(w1eff + (size_t)(i * 32 + lane) * 16);
    const float cb = c1[i * 16 + n16];
    v8f c;
#pragma unroll
    for (int j = 0; j < 8; ++j) c[j] = cb;       // bias broadcast over M rows
    v8f d = __builtin_amdgcn_wmma_f32_16x16x32_f16(false, za, false, bw,
                                                   (short)0, c, false, false);
    _Float16* dst = bufPre + (size_t)(i * 16 + hi * 8) * LDP + n16;
#pragma unroll
    for (int j = 0; j < 8; ++j) dst[j * LDP] = (_Float16)d[j];
  }

  aggregate16(bufPre, bufA, b1, lane);           // relu((h-1+h+h+1)/3 + b1)
  dense_layer(bufA, bufPre, w2lo, w2hi, lane);   // GCN2 pre-agg, K=32(pad)
  aggregate32(bufPre, bufA, b2, lane);
  dense_layer(bufA, bufPre, w3lo, w3hi, lane);   // GCN3 pre-agg

  // ---- final aggregation + b3, f32 coalesced to global (no relu)
  {
    const int f  = (lane & 7) * 4;
    const int nb = lane >> 3;
    const float bv0 = b3[f], bv1 = b3[f + 1], bv2 = b3[f + 2], bv3 = b3[f + 3];
#pragma unroll 2
    for (int k = 0; k < NROWS / 4; ++k) {        // 68 uniform iterations
      const int n  = nb + 4 * k;
      const int i  = n >> 4, g = n & 15;
      const int ip = (i == 0) ? (GNN_NN - 1) : i - 1;
      const int in = (i == GNN_NN - 1) ? 0 : i + 1;
      v4h a = *(const v4h*)(bufPre + n * LDP + f);
      v4h b = *(const v4h*)(bufPre + (ip * 16 + g) * LDP + f);
      v4h c = *(const v4h*)(bufPre + (in * 16 + g) * LDP + f);
      float4 o;
      o.x = ((float)a[0] + (float)b[0] + (float)c[0]) * (1.0f / 3.0f) + bv0;
      o.y = ((float)a[1] + (float)b[1] + (float)c[1]) * (1.0f / 3.0f) + bv1;
      o.z = ((float)a[2] + (float)b[2] + (float)c[2]) * (1.0f / 3.0f) + bv2;
      o.w = ((float)a[3] + (float)b[3] + (float)c[3]) * (1.0f / 3.0f) + bv3;
      *(float4*)(out + ((size_t)(g0 + g) * GNN_NN + i) * 32 + f) = o;
    }
  }
}

// ---------------------------------------------------------------------------
extern "C" void kernel_launch(void* const* d_in, const int* in_sizes, int n_in,
                              void* d_out, int out_size, void* d_ws, size_t ws_size,
                              hipStream_t stream)
{
  (void)n_in; (void)out_size; (void)ws_size;
  const float* z    = (const float*)d_in[0];
  // d_in[1] = edge_index: ring structure is deterministic -> never read.
  const float* fc_w = (const float*)d_in[2];
  const float* fc_b = (const float*)d_in[3];
  const float* w1   = (const float*)d_in[4];
  const float* b1   = (const float*)d_in[5];
  const float* w2   = (const float*)d_in[6];
  const float* b2   = (const float*)d_in[7];
  const float* w3   = (const float*)d_in[8];
  const float* b3   = (const float*)d_in[9];
  float* out = (float*)d_out;

  const int Bg = in_sizes[0] / 32;           // number of graphs (65536)

  char* ws = (char*)d_ws;                    // needs only ~23 KB
  _Float16* w1eff = (_Float16*)(ws + WS_W1EFF);
  _Float16* w2f   = (_Float16*)(ws + WS_W2F);
  _Float16* w3f   = (_Float16*)(ws + WS_W3F);
  float*    c1    = (float*)   (ws + WS_C1);

  hipLaunchKernelGGL(gnn_prep_kernel, dim3(1), dim3(256), 0, stream,
                     fc_w, fc_b, w1, w2, w3, w1eff, w2f, w3f, c1);
  hipLaunchKernelGGL(gnn_main_kernel, dim3(Bg / TILE_G), dim3(32), 0, stream,
                     z, b1, b2, b3, w1eff, w2f, w3f, c1, out);
}
